// AFTModule_60249801228616
// MI455X (gfx1250) — matrix operand order repaired and checked by reading, not compiled, not run on hardware
//
#include <hip/hip_runtime.h>

// AFT forward for MI455X (gfx1250), wave32 WMMA bf16 path.
// T=1024, B=4, D=1024.

#define T_LEN 1024
#define B_SZ  4
#define D_MOD 1024

typedef __attribute__((ext_vector_type(8)))  __bf16 bf16x8;
typedef __attribute__((ext_vector_type(16))) __bf16 bf16x16;
typedef __attribute__((ext_vector_type(8)))  float  floatx8;

static __device__ __forceinline__ __bf16 f2bf(float f) {
  // round-to-nearest-even f32 -> bf16
  unsigned u = __builtin_bit_cast(unsigned, f);
  u += 0x7FFFu + ((u >> 16) & 1u);
  unsigned short s = (unsigned short)(u >> 16);
  return __builtin_bit_cast(__bf16, s);
}

// ---------------------------------------------------------------------------
// Elementwise f32 -> bf16 convert
__global__ __launch_bounds__(256) void cvt_bf16_kernel(
    const float* __restrict__ src, __bf16* __restrict__ dst, int n) {
  int i = blockIdx.x * 256 + threadIdx.x;
  if (i < n) dst[i] = f2bf(src[i]);
}

// exp + convert (for P = exp(pos_bias))
__global__ __launch_bounds__(256) void exp_bf16_kernel(
    const float* __restrict__ src, __bf16* __restrict__ dst, int n) {
  int i = blockIdx.x * 256 + threadIdx.x;
  if (i < n) dst[i] = f2bf(__expf(src[i]));
}

// ---------------------------------------------------------------------------
// LDS-tiled transpose + convert: src [R x C] f32 row-major -> dst [C x R] bf16
// block (32,8), grid (C/32, R/32). R,C multiples of 32.
__global__ __launch_bounds__(256) void transpose_cvt_kernel(
    const float* __restrict__ src, __bf16* __restrict__ dst, int R, int C) {
  __shared__ float tile[32][33];
  int c0 = blockIdx.x * 32, r0 = blockIdx.y * 32;
  for (int i = threadIdx.y; i < 32; i += 8) {
    tile[i][threadIdx.x] = src[(size_t)(r0 + i) * C + (c0 + threadIdx.x)];
  }
  __syncthreads();
  for (int i = threadIdx.y; i < 32; i += 8) {
    dst[(size_t)(c0 + i) * R + (r0 + threadIdx.x)] = f2bf(tile[threadIdx.x][i]);
  }
}

// ---------------------------------------------------------------------------
// Build transposed second-GEMM operand from qkv:
//   Bt2[n][j], n in [0,8192): n = b*1024+d        -> exp(k[j,b,d]) * v[j,b,d]
//                             n = 4096 + b*1024+d -> exp(k[j,b,d])
// qkv is [4096 x 3072] f32 with row m = j*4+b; k at col 1024+d, v at 2048+d.
// block (32,8), grid (T/32, D/32, B).
__global__ __launch_bounds__(256) void prep_b2_kernel(
    const float* __restrict__ qkv, __bf16* __restrict__ bt2) {
  __shared__ float sev[32][33];
  __shared__ float se[32][33];
  int j0 = blockIdx.x * 32, d0 = blockIdx.y * 32, b = blockIdx.z;
  for (int jj = threadIdx.y; jj < 32; jj += 8) {
    const float* row = qkv + (size_t)((j0 + jj) * 4 + b) * 3072;
    float kv = row[1024 + d0 + threadIdx.x];
    float vv = row[2048 + d0 + threadIdx.x];
    float e = __expf(kv);
    sev[jj][threadIdx.x] = e * vv;
    se[jj][threadIdx.x]  = e;
  }
  __syncthreads();
  for (int dd = threadIdx.y; dd < 32; dd += 8) {
    size_t n1 = (size_t)(b * 1024 + d0 + dd);
    int j = j0 + threadIdx.x;
    bt2[n1 * 1024 + j]          = f2bf(sev[threadIdx.x][dd]);
    bt2[(4096 + n1) * 1024 + j] = f2bf(se[threadIdx.x][dd]);
  }
}

// ---------------------------------------------------------------------------
// Gate: y = sigmoid(q) * num / den, stored bf16 [4096 x 1024] (row m = t*4+b).
__global__ __launch_bounds__(256) void gate_kernel(
    const float* __restrict__ qkv, const float* __restrict__ numden,
    __bf16* __restrict__ ybf) {
  int i = blockIdx.x * 256 + threadIdx.x;  // over 4096*1024
  int d = i & 1023;
  int m = i >> 10;      // t*4+b
  int b = m & 3;
  int t = m >> 2;
  float q   = qkv[(size_t)m * 3072 + d];
  float num = numden[(size_t)t * 8192 + b * 1024 + d];
  float den = numden[(size_t)t * 8192 + 4096 + b * 1024 + d];
  float sig = 1.0f / (1.0f + __expf(-q));
  ybf[i] = f2bf(sig * (num / den));
}

// ---------------------------------------------------------------------------
// bf16 WMMA GEMM: C[m][n] = sum_k A[m][k] * Bt[n][k] (+ bias[n])
// A:  [M x K] bf16 row-major, Bt: [N x K] bf16 row-major, C: [M x N] f32.
// Block = 256 threads = 8 waves (2 x 4). Block tile 128x256, wave tile 64x64.
// Per k-step: 16 WMMAs vs 16 global_load_b128 (1:1), acc = 128 VGPRs.
// M mult of 128, N mult of 256, K mult of 32.
template <bool HAS_BIAS>
__global__ __launch_bounds__(256) void gemm_bf16_wmma_kernel(
    const __bf16* __restrict__ A, const __bf16* __restrict__ Bt,
    float* __restrict__ C, const float* __restrict__ bias,
    int M, int N, int K) {
  const int lane = threadIdx.x & 31;
  const int wave = threadIdx.x >> 5;
  const int wm = wave >> 2;          // 0..1
  const int wn = wave & 3;           // 0..3
  const int rowStart = blockIdx.y * 128 + wm * 64;
  const int colStart = blockIdx.x * 256 + wn * 64;
  const int lr = lane & 15;
  const bool hi = lane >= 16;

  floatx8 acc[4][4];
#pragma unroll
  for (int i = 0; i < 4; ++i)
#pragma unroll
    for (int j = 0; j < 4; ++j)
#pragma unroll
      for (int r = 0; r < 8; ++r) acc[i][j][r] = 0.0f;

  const __bf16* pa = A + (size_t)(rowStart + lr) * K + (hi ? 16 : 0);
  const __bf16* pb = Bt + (size_t)(colStart + lr) * K + (hi ? 16 : 0);

  for (int k0 = 0; k0 < K; k0 += 32) {
    bf16x16 afr[4];
    bf16x16 bfr[4];
    // WGP-scope prefetch of this lane's fragment rows two k-iterations ahead
    // (speculative; OOB translation failures are silently dropped).
    __builtin_prefetch(pa + k0 + 64, 0, 3);
    __builtin_prefetch(pb + k0 + 64, 0, 3);
    // A fragment: lanes 0-15 hold K = k0+0..15 (elems 0-15);
    //             lanes 16-31 hold K = k0+16..31.
#pragma unroll
    for (int i = 0; i < 4; ++i) {
      const __bf16* p = pa + (size_t)(16 * i) * K + k0;
      union { bf16x16 v; bf16x8 h[2]; } u;
      u.h[0] = *(const bf16x8*)(p);
      u.h[1] = *(const bf16x8*)(p + 8);
      afr[i] = u.v;
    }
    // B fragment: lane holds column n = colStart+16j+lr, contiguous K run of 16
    // (lanes 0-15: k0..k0+15, lanes 16-31: k0+16..k0+31).
#pragma unroll
    for (int j = 0; j < 4; ++j) {
      bfr[j] = *(const bf16x16*)(pb + (size_t)(16 * j) * K + k0);
    }
#pragma unroll
    for (int i = 0; i < 4; ++i)
#pragma unroll
      for (int j = 0; j < 4; ++j)
        acc[i][j] = __builtin_amdgcn_wmma_f32_16x16x32_bf16(
            false, afr[i], false, bfr[j], (short)0, acc[i][j], false, false);
  }

  // Store: element r of acc maps to row (16i + r + 8*hi), col (16j + lr).
#pragma unroll
  for (int i = 0; i < 4; ++i) {
#pragma unroll
    for (int j = 0; j < 4; ++j) {
      const int col = colStart + 16 * j + lr;
      float bv = 0.0f;
      if (HAS_BIAS) bv = bias[col];
#pragma unroll
      for (int r = 0; r < 8; ++r) {
        const int row = rowStart + 16 * i + r + (hi ? 8 : 0);
        C[(size_t)row * N + col] = acc[i][j][r] + bv;
      }
    }
  }
}

// ---------------------------------------------------------------------------
extern "C" void kernel_launch(void* const* d_in, const int* in_sizes, int n_in,
                              void* d_out, int out_size, void* d_ws, size_t ws_size,
                              hipStream_t stream) {
  const float* data     = (const float*)d_in[0];  // [T,B,D]
  const float* W_qkv    = (const float*)d_in[1];  // [D,3D]
  const float* b_qkv    = (const float*)d_in[2];  // [3D]
  const float* pos_bias = (const float*)d_in[3];  // [T,T]
  const float* W_out    = (const float*)d_in[4];  // [D,D]
  const float* b_out    = (const float*)d_in[5];  // [D]
  float* out = (float*)d_out;                     // [T,B,D]

  unsigned char* ws = (unsigned char*)d_ws;
  // Workspace layout (bytes), buffers overlaid by lifetime. Peak = 112 MB.
  const size_t OFF_QKV = 0;                      // 4096*3072*4 = 48 MB (f32)
  const size_t OFF_XBF = 50331648;               // 4096*1024*2 =  8 MB (bf16)
  const size_t OFF_WT  = 58720256;               // 3072*1024*2 =  6 MB (bf16)
  const size_t OFF_PBF = 65011712;               // 1024*1024*2 =  2 MB (bf16)
  const size_t OFF_BT2 = 67108864;               // 8192*1024*2 = 16 MB (bf16)
  const size_t OFF_ND  = 83886080;               // 1024*8192*4 = 32 MB (f32)
  const size_t OFF_YBF = OFF_XBF;                // reuse (Xbf dead after GEMM1)
  const size_t OFF_WOT = OFF_WT;                 // reuse (Wqkvt dead after GEMM1)
  (void)ws_size; (void)in_sizes; (void)n_in; (void)out_size;

  float*  qkv   = (float*)(ws + OFF_QKV);
  __bf16* Xbf   = (__bf16*)(ws + OFF_XBF);
  __bf16* Wqkvt = (__bf16*)(ws + OFF_WT);
  __bf16* Pbf   = (__bf16*)(ws + OFF_PBF);
  __bf16* Bt2   = (__bf16*)(ws + OFF_BT2);
  float*  numden= (float*)(ws + OFF_ND);
  __bf16* Ybf   = (__bf16*)(ws + OFF_YBF);
  __bf16* Wot   = (__bf16*)(ws + OFF_WOT);

  const int MB = T_LEN * B_SZ;   // 4096
  const int TD = MB * D_MOD;     // 4 M elements

  // Stage 0: bf16 copies (A row-major; B operands transposed to [N x K]).
  cvt_bf16_kernel<<<TD / 256, 256, 0, stream>>>(data, Xbf, TD);
  transpose_cvt_kernel<<<dim3(3 * D_MOD / 32, D_MOD / 32), dim3(32, 8), 0, stream>>>(
      W_qkv, Wqkvt, D_MOD, 3 * D_MOD);

  // Stage 1: qkv = data @ W_qkv + b_qkv    (M=4096, N=3072, K=1024)
  gemm_bf16_wmma_kernel<true><<<dim3(3 * D_MOD / 256, MB / 128), 256, 0, stream>>>(
      Xbf, Wqkvt, qkv, b_qkv, MB, 3 * D_MOD, D_MOD);

  // Stage 2: P = exp(pos_bias) in bf16; Bt2 = [ek*v ; ek] transposed to [8192 x 1024].
  exp_bf16_kernel<<<(T_LEN * T_LEN) / 256, 256, 0, stream>>>(pos_bias, Pbf, T_LEN * T_LEN);
  prep_b2_kernel<<<dim3(T_LEN / 32, D_MOD / 32, B_SZ), dim3(32, 8), 0, stream>>>(qkv, Bt2);

  // Stage 3: [num | den] = P @ [ekv | ek]  (M=1024, N=8192, K=1024)
  gemm_bf16_wmma_kernel<false><<<dim3(8192 / 256, T_LEN / 128), 256, 0, stream>>>(
      Pbf, Bt2, numden, nullptr, T_LEN, 8192, T_LEN);

  // Stage 4: y = sigmoid(q) * num/den (bf16), transpose-convert W_out.
  gate_kernel<<<TD / 256, 256, 0, stream>>>(qkv, numden, Ybf);
  transpose_cvt_kernel<<<dim3(D_MOD / 32, D_MOD / 32), dim3(32, 8), 0, stream>>>(
      W_out, Wot, D_MOD, D_MOD);

  // Stage 5: out = y @ W_out + b_out       (M=4096, N=1024, K=1024)
  gemm_bf16_wmma_kernel<true><<<dim3(D_MOD / 256, MB / 128), 256, 0, stream>>>(
      Ybf, Wot, out, b_out, MB, D_MOD, D_MOD);
}
